// CoSSM_120259084379
// MI455X (gfx1250) — matrix operand order repaired and checked by reading, not compiled, not run on hardware
//
#include <hip/hip_runtime.h>
#include <cstdint>

// CoSSM (CNN + bidirectional Mamba) forward for MI455X / gfx1250.
// All matmul work (conv3, skip 1x1, in/x/dt/out projections) runs through
// v_wmma_f32_16x16x32_bf16 (f32->bf16 A/B, f32 accum) with 2x2 register
// blocking (32x32 output per wave) and unconditional float4
// (global_load_b128) fragment loads. The serial selective scan is VALU with
// B/C staged into LDS via global_load_async_to_lds_b32.

typedef __attribute__((ext_vector_type(16))) __bf16 v16bf;
typedef __attribute__((ext_vector_type(8)))  float  v8f;

#define WPB 8   // waves per block in WMMA kernels (8 * 32 = 256 threads)

__device__ __forceinline__ float sigmoidf_(float x) { return 1.f / (1.f + __expf(-x)); }

// ---- CDNA5 async global->LDS copy (ASYNCcnt path) -------------------------
__device__ __forceinline__ void async_b32(uint32_t lds_off, const void* gaddr) {
  asm volatile("global_load_async_to_lds_b32 %0, %1, off"
               :: "v"(lds_off), "v"(gaddr) : "memory");
}
__device__ __forceinline__ void wait_async0() {
  asm volatile("s_wait_asynccnt 0" ::: "memory");
}

// pack 4 f32 into 4 bf16 lanes of a fragment (constant base after unroll)
__device__ __forceinline__ void cvt4(v16bf& v, int base, float4 f) {
  v[base + 0] = (__bf16)f.x;
  v[base + 1] = (__bf16)f.y;
  v[base + 2] = (__bf16)f.z;
  v[base + 3] = (__bf16)f.w;
}

// A fragment (16x32 bf16): lane covers row m0+(lane&15); per lane the K values
// are one contiguous 16-float run at kbase = (lane>>4)*16.
__device__ __forceinline__ v16bf load_a_frag(const float* __restrict__ p) {
  const float4* q0 = (const float4*)p;         // 8 floats
  const float4* q1 = (const float4*)(p + 8);   // 8 floats
  v16bf f;
  cvt4(f, 0, q0[0]); cvt4(f, 4, q0[1]);
  cvt4(f, 8, q1[0]); cvt4(f, 12, q1[1]);
  return f;
}

// B fragment (32x16 bf16): lane covers col n0+(lane&15); K values are one
// contiguous 16-float run at (lane>>4)*16.
__device__ __forceinline__ v16bf load_b_frag(const float* __restrict__ p) {
  const float4* q = (const float4*)p;
  v16bf f;
  cvt4(f, 0, q[0]); cvt4(f, 4, q[1]); cvt4(f, 8, q[2]); cvt4(f, 12, q[3]);
  return f;
}

__device__ __forceinline__ v8f wmma_bf16(v16bf a, v16bf b, v8f c) {
  return __builtin_amdgcn_wmma_f32_16x16x32_bf16(false, a, false, b,
                                                 (short)0, c, false, false);
}

// ---------------------------------------------------------------------------
// Generic GEMM: C[M,N] = epi(A[M,K] @ W[N,K]^T + bias[N])
// Requirements: K % 32 == 0, M % 32 == 0, N % 16 == 0.
// epi: 0 = none, 1 = SiLU, 2 = softplus.
// One wave computes a 32x32 supertile (2x2 WMMA register blocking); a
// trailing odd N-tile is handled with a wave-uniform guard.
// ---------------------------------------------------------------------------
__global__ void gemm_wmma_kernel(const float* __restrict__ A, int lda, int K,
                                 const float* __restrict__ W, int ldw,
                                 const float* __restrict__ bias,
                                 float* __restrict__ C, int ldc,
                                 int M, int N, int epi) {
  const int lane = threadIdx.x & 31;
  const int wave = threadIdx.x >> 5;
  const int stM = M >> 5;                 // supertile rows (32 wide)
  const int stN = ((N >> 4) + 1) >> 1;    // supertile cols (2 N-tiles each)
  int tile = blockIdx.x * WPB + wave;
  if (tile >= stM * stN) return;
  const int m0 = (tile / stN) << 5;
  const int n0 = (tile % stN) << 5;
  const bool has_n1 = (n0 + 16) < N;      // wave-uniform

  const float* __restrict__ arow0 =
      A + (size_t)(m0 + (lane & 15)) * lda + ((lane >> 4) << 4);
  const float* __restrict__ arow1 = arow0 + (size_t)16 * lda;
  const float* __restrict__ wrow0 =
      W + (size_t)(n0 + (lane & 15)) * ldw + ((lane >> 4) << 4);
  const float* __restrict__ wrow1 = wrow0 + (size_t)16 * ldw;

  v8f acc00 = {}, acc01 = {}, acc10 = {}, acc11 = {};
  for (int k0 = 0; k0 < K; k0 += 32) {
    __builtin_prefetch(arow0 + k0 + 64, 0, 0);
    v16bf a0 = load_a_frag(arow0 + k0);
    v16bf a1 = load_a_frag(arow1 + k0);
    v16bf b0 = load_b_frag(wrow0 + k0);
    acc00 = wmma_bf16(a0, b0, acc00);
    acc10 = wmma_bf16(a1, b0, acc10);
    if (has_n1) {
      v16bf b1 = load_b_frag(wrow1 + k0);
      acc01 = wmma_bf16(a0, b1, acc01);
      acc11 = wmma_bf16(a1, b1, acc11);
    }
  }

  auto store_tile = [&](v8f acc, int mb, int nn) {
    const float bv = bias ? bias[nn] : 0.f;
#pragma unroll
    for (int r = 0; r < 8; ++r) {
      float v = acc[r] + bv;
      if (epi == 1)      v = v * sigmoidf_(v);                         // SiLU
      else if (epi == 2) v = (v > 20.f) ? v : __logf(1.f + __expf(v)); // softplus
      C[(size_t)(mb + r) * ldc + nn] = v;
    }
  };
  const int mb0 = m0 + ((lane >> 4) << 3);
  const int nn0 = n0 + (lane & 15);
  store_tile(acc00, mb0, nn0);
  store_tile(acc10, mb0 + 16, nn0);
  if (has_n1) {
    store_tile(acc01, mb0, nn0 + 16);
    store_tile(acc11, mb0 + 16, nn0 + 16);
  }
}

// ---------------------------------------------------------------------------
// CNN stream: Y[t, co] = relu(BN(sum_{k,cin} X[t+k-1, cin] * Wc[co,cin,k]))
//                        + (Wskip ? X @ Wskip^T : X)     (token-major layout)
// Wc: [Cout, Cin, 3]; Wskip: [Cout, Cin] or null. Out-of-range taps read the
// zeroed scratch row `zrow` so all fragment loads are unconditional.
// One wave covers a 32x16 output tile (2x M-blocking) so both row blocks
// share the expensive stride-3 weight gathers.
// ---------------------------------------------------------------------------
__global__ void conv3_bn_relu_skip_kernel(const float* __restrict__ X, int Cin,
                                          const float* __restrict__ Wc,
                                          const float* __restrict__ bn_s,
                                          const float* __restrict__ bn_b,
                                          const float* __restrict__ bn_m,
                                          const float* __restrict__ bn_v,
                                          const float* __restrict__ Wskip,
                                          const float* __restrict__ zrow,
                                          float* __restrict__ Y, int Cout,
                                          int L, int M) {
  const int lane = threadIdx.x & 31;
  const int wave = threadIdx.x >> 5;
  const int tilesN = Cout >> 4;
  const int stM = M >> 5;
  int tile = blockIdx.x * WPB + wave;
  if (tile >= stM * tilesN) return;
  const int m0 = (tile / tilesN) << 5;
  const int n0 = (tile % tilesN) << 4;

  const int ka_base = (lane >> 4) << 4;
  const int kb_base = (lane >> 4) << 4;
  const float* ar[2][3];
#pragma unroll
  for (int mb = 0; mb < 2; ++mb) {
    int t = m0 + mb * 16 + (lane & 15);
    int b = t / L, l = t % L;
#pragma unroll
    for (int k = 0; k < 3; ++k) {           // taps at l-1, l, l+1 (zero pad)
      int sl = l + k - 1;
      bool ok = (sl >= 0) && (sl < L);
      ar[mb][k] = (ok ? (X + (size_t)(b * L + sl) * Cin) : zrow) + ka_base;
    }
  }
  const int nB = n0 + (lane & 15);
  const float* wbase = Wc + (size_t)nB * Cin * 3;
  const float* wskiprow = Wskip ? (Wskip + (size_t)nB * Cin + kb_base) : nullptr;

  v8f acc[2] = {{}, {}}, accs[2] = {{}, {}};
  for (int c0 = 0; c0 < Cin; c0 += 32) {    // Cin is a multiple of 32
    v16bf af[2][3];
#pragma unroll
    for (int mb = 0; mb < 2; ++mb)
#pragma unroll
      for (int k = 0; k < 3; ++k) af[mb][k] = load_a_frag(ar[mb][k] + c0);
#pragma unroll
    for (int k = 0; k < 3; ++k) {
      v16bf bf;                             // stride-3 weight gather
#pragma unroll
      for (int i = 0; i < 16; ++i)
        bf[i] = (__bf16)wbase[(size_t)(c0 + kb_base + i) * 3 + k];
#pragma unroll
      for (int mb = 0; mb < 2; ++mb)
        acc[mb] = wmma_bf16(af[mb][k], bf, acc[mb]);
    }
    if (wskiprow) {                         // 1x1 skip conv, unshifted rows
      v16bf bs = load_b_frag(wskiprow + c0);
#pragma unroll
      for (int mb = 0; mb < 2; ++mb)
        accs[mb] = wmma_bf16(af[mb][1], bs, accs[mb]);
    }
  }
  const float s  = bn_s[nB] * rsqrtf(bn_v[nB] + 1e-5f);
  const float sh = bn_b[nB] - bn_m[nB] * s;
#pragma unroll
  for (int mb = 0; mb < 2; ++mb) {
    const int mbase = m0 + mb * 16 + ((lane >> 4) << 3);
#pragma unroll
    for (int r = 0; r < 8; ++r) {
      int tt = mbase + r;
      float cv = fmaxf(acc[mb][r] * s + sh, 0.f);
      float sk = Wskip ? accs[mb][r] : X[(size_t)tt * Cin + nB]; // identity: Cin==Cout
      Y[(size_t)tt * Cout + nB] = cv + sk;
    }
  }
}

// ---------------------------------------------------------------------------
// Causal depthwise conv (K=4) + bias + SiLU. rev=1 operates on the reversed
// sequence and writes u in logical (reversed) order.
// ---------------------------------------------------------------------------
__global__ void dwconv_silu_kernel(const float* __restrict__ x,
                                   const float* __restrict__ w,   // [di,1,4]
                                   const float* __restrict__ bias,
                                   float* __restrict__ u,
                                   int L, int di, int rev, long total) {
  long i = (long)blockIdx.x * blockDim.x + threadIdx.x;
  if (i >= total) return;
  int c = (int)(i % di);
  long bp = i / di;
  int p = (int)(bp % L);
  int b = (int)(bp / L);
  float s = bias[c];
#pragma unroll
  for (int j = 0; j < 4; ++j) {
    int q = p - 3 + j;
    if (q >= 0) {
      int phys = rev ? (L - 1 - q) : q;
      s += x[((size_t)b * L + phys) * di + c] * w[c * 4 + j];
    }
  }
  u[((size_t)b * L + p) * di + c] = s * sigmoidf_(s);
}

// ---------------------------------------------------------------------------
// Selective scan: h[n] = exp(dt*A[c,n])*h[n] + dt*u*B[l,n]; y = h.C[l] + D*u
// B/C live in xdbl[:, bc_off : bc_off+32]; staged 16 timesteps at a time into
// LDS with async global->LDS loads. One thread per (b, channel c).
// ---------------------------------------------------------------------------
__global__ void scan_kernel(const float* __restrict__ u,
                            const float* __restrict__ dtv,
                            const float* __restrict__ xdbl, int ldx, int bc_off,
                            const float* __restrict__ Alog,
                            const float* __restrict__ Dp,
                            float* __restrict__ y,
                            int L, int di) {
  __shared__ float sB[256];   // [16 steps][16 states]
  __shared__ float sC[256];
  const int tid = threadIdx.x;
  const int c = blockIdx.x * 256 + tid;
  const int b = blockIdx.y;
  float a_[16], h[16];
#pragma unroll
  for (int n = 0; n < 16; ++n) {
    a_[n] = -__expf(Alog[(size_t)c * 16 + n]);
    h[n] = 0.f;
  }
  const float dD = Dp[c];
  const int s = tid >> 4, j = tid & 15;               // tid == s*16 + j
  const uint32_t offB = (uint32_t)(uintptr_t)&sB[tid];
  const uint32_t offC = (uint32_t)(uintptr_t)&sC[tid];
  const size_t base_bl = (size_t)b * L;

  for (int l0 = 0; l0 < L; l0 += 16) {
    const float* gB = xdbl + (base_bl + l0 + s) * (size_t)ldx + bc_off + j;
    __syncthreads();                 // previous chunk fully consumed
    async_b32(offB, gB);             // B[l0+s, j]
    async_b32(offC, gB + 16);        // C[l0+s, j]
    wait_async0();
    __syncthreads();                 // all waves' LDS data visible
#pragma unroll 2
    for (int st = 0; st < 16; ++st) {
      size_t idx = (base_bl + l0 + st) * (size_t)di + c;
      float dt_t = dtv[idx];
      float u_t  = u[idx];
      float du   = dt_t * u_t;
      float yacc = 0.f;
#pragma unroll
      for (int n = 0; n < 16; ++n) {
        h[n] = __expf(dt_t * a_[n]) * h[n] + du * sB[st * 16 + n];
        yacc += h[n] * sC[st * 16 + n];
      }
      y[idx] = yacc + dD * u_t;
    }
  }
}

// ---------------------------------------------------------------------------
// g = (yf + reverse_l(yb)) * silu(z)
// ---------------------------------------------------------------------------
__global__ void combine_gate_kernel(const float* __restrict__ yf,
                                    const float* __restrict__ yb,
                                    const float* __restrict__ z,
                                    float* __restrict__ g,
                                    int L, int di, long total) {
  long i = (long)blockIdx.x * blockDim.x + threadIdx.x;
  if (i >= total) return;
  int c = (int)(i % di);
  long bp = i / di;
  int p = (int)(bp % L);
  int b = (int)(bp / L);
  float yv = yf[i] + yb[((size_t)b * L + (L - 1 - p)) * di + c];
  float zv = z[i];
  g[i] = yv * (zv * sigmoidf_(zv));
}

// ---------------------------------------------------------------------------
// out[t,:] = resid[t,:] + LayerNorm(yp[t,:]; g, b)   (one block per token)
// ---------------------------------------------------------------------------
__global__ void ln_residual_kernel(const float* __restrict__ yp,
                                   const float* __restrict__ resid,
                                   const float* __restrict__ g,
                                   const float* __restrict__ bta,
                                   float* __restrict__ out, int d) {
  __shared__ float red[256];
  const int t = blockIdx.x;
  const int tid = threadIdx.x;
  const float* row = yp + (size_t)t * d;
  float s = 0.f;
  for (int i = tid; i < d; i += 256) s += row[i];
  red[tid] = s; __syncthreads();
  for (int o = 128; o > 0; o >>= 1) { if (tid < o) red[tid] += red[tid + o]; __syncthreads(); }
  const float mu = red[0] / d; __syncthreads();
  float v = 0.f;
  for (int i = tid; i < d; i += 256) { float dd = row[i] - mu; v += dd * dd; }
  red[tid] = v; __syncthreads();
  for (int o = 128; o > 0; o >>= 1) { if (tid < o) red[tid] += red[tid + o]; __syncthreads(); }
  const float inv = rsqrtf(red[0] / d + 1e-6f);
  for (int i = tid; i < d; i += 256)
    out[(size_t)t * d + i] = resid[(size_t)t * d + i] + (row[i] - mu) * inv * g[i] + bta[i];
}

// ---------------------------------------------------------------------------
// small helpers: zero a scratch row; pad dt_proj_w [rows,kin] -> [rows,32]
// ---------------------------------------------------------------------------
__global__ void zero_kernel(float* __restrict__ p, int n) {
  int i = blockIdx.x * blockDim.x + threadIdx.x;
  if (i < n) p[i] = 0.f;
}
__global__ void pad_w_kernel(const float* __restrict__ src, float* __restrict__ dst,
                             int rows, int kin, int kout) {
  long i = (long)blockIdx.x * blockDim.x + threadIdx.x;
  if (i >= (long)rows * kout) return;
  int r = (int)(i / kout), k = (int)(i % kout);
  dst[i] = (k < kin) ? src[(size_t)r * kin + k] : 0.f;
}

// ---------------------------------------------------------------------------
extern "C" void kernel_launch(void* const* d_in, const int* in_sizes, int n_in,
                              void* d_out, int out_size, void* d_ws, size_t ws_size,
                              hipStream_t stream) {
  (void)in_sizes; (void)n_in; (void)out_size; (void)ws_size;
  const int Bb = 4, L = 1024, T = Bb * L;
  const int sizes[3] = {256, 512, 512};

  int ii = 0;
  auto F = [&](void) { return (const float*)d_in[ii++]; };
  const float* a_x = F();
  const float* v_x = F();

  struct Blk {
    const float *conv_w[2], *bn_s[2], *bn_b[2], *bn_m[2], *bn_v[2], *skip_w[2];
    const float *in_proj;
    const float *dw_w[2], *dw_b[2], *xproj[2], *dtw[2], *dtbias[2], *Alog[2], *Dd[2];
    const float *out_proj, *ln_g[2], *ln_b[2];
  } P[3];

  int prev = 128;
  for (int i = 0; i < 3; ++i) {
    int d = sizes[i];
    bool skip = (prev != d);
    for (int m = 0; m < 2; ++m) {   // 'a' stream then 'v' stream
      P[i].conv_w[m] = F(); P[i].bn_s[m] = F(); P[i].bn_b[m] = F();
      P[i].bn_m[m] = F();  P[i].bn_v[m] = F();
      P[i].skip_w[m] = skip ? F() : nullptr;
    }
    P[i].in_proj = F();
    for (int s = 0; s < 2; ++s) {   // fwd then bwd scan params
      P[i].dw_w[s] = F(); P[i].dw_b[s] = F(); P[i].xproj[s] = F();
      P[i].dtw[s] = F();  P[i].dtbias[s] = F(); P[i].Alog[s] = F(); P[i].Dd[s] = F();
    }
    P[i].out_proj = F();
    P[i].ln_g[0] = F(); P[i].ln_b[0] = F();   // ln1 -> stream a
    P[i].ln_g[1] = F(); P[i].ln_b[1] = F();   // ln2 -> stream v
    prev = d;
  }

  // Workspace layout (~135 MB of f32)
  float* ws = (float*)d_ws;
  size_t off = 0;
  auto alloc = [&](size_t n) { float* p = ws + off; off += n; return p; };
  float* buf[2] = { alloc((size_t)T * 512), alloc((size_t)T * 512) };
  float* tmp   = alloc((size_t)T * 512);    // CNN output / residual
  float* yproj = alloc((size_t)T * 512);
  float* xin   = alloc((size_t)T * 1024);
  float* zb    = alloc((size_t)T * 1024);
  float* ub    = alloc((size_t)T * 1024);   // also reused as gate buffer
  float* dtb   = alloc((size_t)T * 1024);
  float* yfb   = alloc((size_t)T * 1024);
  float* ybb   = alloc((size_t)T * 1024);
  float* xdbl  = alloc((size_t)T * 64);
  float* zrow  = alloc(512);                // zero page for conv edge taps
  float* dtwp  = alloc((size_t)1024 * 32);  // dt_proj_w padded to K=32
  float* gb = ub;

  zero_kernel<<<dim3(2), dim3(256), 0, stream>>>(zrow, 512);

  auto gemm = [&](const float* A, int lda, int K, const float* W, int ldw,
                  const float* bias, float* Cc, int ldc, int M, int N, int epi) {
    int stiles = (M / 32) * (((N / 16) + 1) / 2);
    gemm_wmma_kernel<<<dim3((stiles + WPB - 1) / WPB), dim3(32 * WPB), 0, stream>>>(
        A, lda, K, W, ldw, bias, Cc, ldc, M, N, epi);
  };

  prev = 128;
  for (int blk = 0; blk < 3; ++blk) {
    const int d = sizes[blk], di = 2 * d, dtr = d / 16, ldx = dtr + 32;
    const long tot_di = (long)T * di;
    const int ew_blocks = (int)((tot_di + 255) / 256);

    for (int st = 0; st < 2; ++st) {
      const float* xsrc = (blk == 0) ? (st ? v_x : a_x) : buf[st];

      // 1) CNN stream (conv3 + BN + ReLU + skip) -> tmp [T, d]
      {
        int tiles = (T / 32) * (d / 16);
        conv3_bn_relu_skip_kernel<<<dim3((tiles + WPB - 1) / WPB), dim3(32 * WPB), 0, stream>>>(
            xsrc, prev, P[blk].conv_w[st], P[blk].bn_s[st], P[blk].bn_b[st],
            P[blk].bn_m[st], P[blk].bn_v[st], P[blk].skip_w[st], zrow, tmp, d, L, T);
      }

      // 2) in_proj: xin = tmp @ W[0:di]^T ; z = tmp @ W[di:2di]^T
      gemm(tmp, d, d, P[blk].in_proj, d, nullptr, xin, di, T, di, 0);
      gemm(tmp, d, d, P[blk].in_proj + (size_t)di * d, d, nullptr, zb, di, T, di, 0);

      // 3) forward / backward scans
      for (int dir = 0; dir < 2; ++dir) {
        dwconv_silu_kernel<<<dim3(ew_blocks), dim3(256), 0, stream>>>(
            xin, P[blk].dw_w[dir], P[blk].dw_b[dir], ub, L, di, dir, tot_di);
        // x_proj -> [T, dtr+32] (dt_r | B | C)
        gemm(ub, di, di, P[blk].xproj[dir], di, nullptr, xdbl, ldx, T, ldx, 0);
        // dt = softplus(xdbl[:, :dtr] @ dt_proj^T + bias); weights padded so
        // that K=32 with zero weights over the B columns of xdbl.
        pad_w_kernel<<<dim3((di * 32 + 255) / 256), dim3(256), 0, stream>>>(
            P[blk].dtw[dir], dtwp, di, dtr, 32);
        gemm(xdbl, ldx, 32, dtwp, 32, P[blk].dtbias[dir], dtb, di, T, di, 2);
        // selective scan
        scan_kernel<<<dim3(di / 256, Bb), dim3(256), 0, stream>>>(
            ub, dtb, xdbl, ldx, dtr, P[blk].Alog[dir], P[blk].Dd[dir],
            dir ? ybb : yfb, L, di);
      }

      // 4) gate + combine (un-reverses backward output)
      combine_gate_kernel<<<dim3(ew_blocks), dim3(256), 0, stream>>>(
          yfb, ybb, zb, gb, L, di, tot_di);

      // 5) out_proj -> yproj [T, d]
      gemm(gb, di, di, P[blk].out_proj, di, nullptr, yproj, d, T, d, 0);

      // 6) residual + LayerNorm
      float* dst = (blk == 2) ? ((float*)d_out + (size_t)st * T * 512) : buf[st];
      ln_residual_kernel<<<dim3(T), dim3(256), 0, stream>>>(
          yproj, tmp, P[blk].ln_g[st], P[blk].ln_b[st], dst, d);
    }
    prev = d;
  }
}
